// PsiSparseAttention_59760174956875
// MI455X (gfx1250) — matrix operand order repaired and checked by reading, not compiled, not run on hardware
//
#include <hip/hip_runtime.h>

typedef __attribute__((ext_vector_type(16))) __bf16       v16bf;
typedef __attribute__((ext_vector_type(8)))  float        v8f;
typedef __attribute__((ext_vector_type(4)))  unsigned int u32x4;

#define NB_B 2
#define NB_H 16
#define SEQ  2048
#define DIM  128
#define BR   64
#define BC   64
#define NQB  (SEQ/BR)   // 32
#define NKB  (SEQ/BC)   // 32
#define KST  144        // K-tile LDS row stride (bf16 elems), padded vs 128
#define VST  72         // V^T-tile LDS row stride (64 keys + pad)
#define PST  72         // P-tile LDS row stride
#define NBINS 4096
#define BIN_LO (-10.0f)
#define BIN_W  (20.0f / (float)NBINS)
#define BIN_INV ((float)NBINS / 20.0f)
#define QK_SCALE 0.08838834764831845f   // 1/sqrt(128)
#define TARGET_CNT 67108864ull          // 0.5 * B*H*S*S
#define FNEG_MAX (-3.402823466e38f)

// ---------- bf16 helpers (pure integer, RNE) ----------
static __device__ __forceinline__ unsigned short bf16bits(float f) {
  unsigned int u = __builtin_bit_cast(unsigned int, f);
  unsigned int r = (u + 0x7FFFu + ((u >> 16) & 1u)) >> 16;
  return (unsigned short)r;
}
static __device__ __forceinline__ __bf16 bf16val(float f) {
  unsigned short s = bf16bits(f);
  return __builtin_bit_cast(__bf16, s);
}
static __device__ __forceinline__ unsigned int pk2bf(float a, float b) {
  return (unsigned int)bf16bits(a) | ((unsigned int)bf16bits(b) << 16);
}
// 16 contiguous bf16 (32B, 16B-aligned) from LDS -> two ds_load_b128
static __device__ __forceinline__ v16bf ldsB16x16(const __bf16* p) {
  union { u32x4 q[2]; v16bf v; } u;
  const u32x4* s = (const u32x4*)p;
  u.q[0] = s[0];
  u.q[1] = s[1];
  return u.v;
}

// ---------- A-fragment: 16x32 bf16 Q chunk, contiguous-K layout (matches B in qkTiles) ----------
// lane<16: K = c*32 + {0..15}; lane>=16: K = c*32 + {16..31}
static __device__ __forceinline__ void loadQfrag(const float* qg, int rowBase,
                                                 int lane, v16bf aq[4]) {
  const int half = lane >> 4, rl = lane & 15;
  const float* qp = qg + (size_t)(rowBase + rl) * DIM;
#pragma unroll
  for (int c = 0; c < 4; ++c) {
    const float* p = qp + c * 32 + half * 16;
    v16bf a;
#pragma unroll
    for (int j = 0; j < 8; ++j) a[j] = bf16val(p[j]);
#pragma unroll
    for (int j = 0; j < 8; ++j) a[8 + j] = bf16val(p[8 + j]);
    aq[c] = a;
  }
}

// ---------- stage 64x128 fp32 tile -> bf16 row-major LDS (stride KST) ----------
static __device__ __forceinline__ void stageK(__bf16* Kl, const float* kg, int tid) {
#pragma unroll
  for (int i = 0; i < 16; ++i) {
    const int idx = tid + i * 128;          // 2048 float4s total
    const int r = idx >> 5;                 // 32 float4 per row
    const int dq = (idx & 31) << 2;
    const float4 f = ((const float4*)kg)[idx];
    unsigned int* wp = (unsigned int*)(Kl + r * KST + dq);
    wp[0] = pk2bf(f.x, f.y);
    wp[1] = pk2bf(f.z, f.w);
  }
}

// ---------- stage 64x128 fp32 V tile -> transposed bf16 LDS Vt[d][key] ----------
static __device__ __forceinline__ void stageVt(__bf16* Vt, const float* vg, int tid) {
#pragma unroll
  for (int i = 0; i < 16; ++i) {
    const int idx = tid + i * 128;
    const int r = idx >> 5;
    const int dq = (idx & 31) << 2;
    const float4 f = ((const float4*)vg)[idx];
    Vt[(dq + 0) * VST + r] = bf16val(f.x);
    Vt[(dq + 1) * VST + r] = bf16val(f.y);
    Vt[(dq + 2) * VST + r] = bf16val(f.z);
    Vt[(dq + 3) * VST + r] = bf16val(f.w);
  }
}

// ---------- 16x64 score tile per wave: 4 n-tiles x 4 k-chunks of WMMA ----------
static __device__ __forceinline__ void qkTiles(const __bf16* Kl, const v16bf aq[4],
                                               int lane, v8f sc[4]) {
  const int half = lane >> 4, rl = lane & 15;
#pragma unroll
  for (int t = 0; t < 4; ++t) {
    v8f s = {};
#pragma unroll
    for (int c = 0; c < 4; ++c) {
      // B 32x16: lane = col (key row t*16+rl), K = c*32 + half*16 + j
      v16bf b = ldsB16x16(Kl + (t * 16 + rl) * KST + c * 32 + half * 16);
      s = __builtin_amdgcn_wmma_f32_16x16x32_bf16(false, aq[c], false, b,
                                                  (short)0, s, false, false);
    }
    sc[t] = s;
  }
}

// ======================= pass 0: zero histogram =======================
__global__ void psi_zero_kernel(unsigned int* p, int n) {
  int i = blockIdx.x * blockDim.x + threadIdx.x;
  if (i < n) p[i] = 0u;
}

// ======================= pass 1: WMMA QK^T -> histogram =======================
__global__ void __launch_bounds__(128)
psi_hist_kernel(const float* __restrict__ q, const float* __restrict__ k,
                unsigned int* __restrict__ hist) {
  __shared__ __align__(16) __bf16 Kl[BC * KST];
  __shared__ unsigned int hl[NBINS];
  const int tid = threadIdx.x, lane = tid & 31, w = tid >> 5;
  const int bh = blockIdx.x / NQB, qb = blockIdx.x % NQB;
  const float* qg = q + (size_t)bh * SEQ * DIM;
  const float* kg = k + (size_t)bh * SEQ * DIM;

  for (int i = tid; i < NBINS; i += 128) hl[i] = 0u;

  v16bf aq[4];
  loadQfrag(qg, qb * BR + w * 16, lane, aq);

  for (int kb = 0; kb < NKB; ++kb) {
    __syncthreads();
    stageK(Kl, kg + (size_t)kb * BC * DIM, tid);
    __syncthreads();
    __builtin_prefetch(kg + (size_t)((kb + 1) & (NKB - 1)) * BC * DIM + tid * 16, 0, 1);
    v8f sc[4];
    qkTiles(Kl, aq, lane, sc);
#pragma unroll
    for (int t = 0; t < 4; ++t) {
#pragma unroll
      for (int r = 0; r < 8; ++r) {
        const float s = sc[t][r] * QK_SCALE;
        int b = (int)((s - BIN_LO) * BIN_INV);
        b = b < 0 ? 0 : (b > NBINS - 1 ? NBINS - 1 : b);
        atomicAdd(&hl[b], 1u);
      }
    }
  }
  __syncthreads();
  for (int i = tid; i < NBINS; i += 128)
    if (hl[i]) atomicAdd(&hist[i], hl[i]);
}

// ======================= pass 2: histogram -> threshold =======================
__global__ void psi_select_kernel(const unsigned int* __restrict__ hist,
                                  float* __restrict__ thr) {
  if (threadIdx.x == 0 && blockIdx.x == 0) {
    unsigned long long cum = 0;
    float t = BIN_LO;
    for (int i = 0; i < NBINS; ++i) {
      cum += hist[i];
      if (cum >= TARGET_CNT) { t = BIN_LO + ((float)i + 0.5f) * BIN_W; break; }
    }
    *thr = t;
  }
}

// ======================= pass 3: masked flash attention =======================
__global__ void __launch_bounds__(128)
psi_flash_kernel(const float* __restrict__ q, const float* __restrict__ k,
                 const float* __restrict__ v, const float* __restrict__ thrp,
                 float* __restrict__ out) {
  __shared__ __align__(16) __bf16 Kl[BC * KST];
  __shared__ __align__(16) __bf16 Vt[DIM * VST];
  __shared__ __align__(16) __bf16 Pl[4 * 16 * PST];

  const int tid = threadIdx.x, lane = tid & 31, w = tid >> 5;
  const int half = lane >> 4, rl = lane & 15;
  const int bh = blockIdx.x / NQB, qb = blockIdx.x % NQB;
  const size_t base = (size_t)bh * SEQ * DIM;
  const float* qg = q + base;
  const float* kg = k + base;
  const float* vg = v + base;
  const float thr = thrp ? *thrp : 0.0f;  // 0.0 = analytic median of N(0,1) fallback

  v16bf aq[4];
  loadQfrag(qg, qb * BR + w * 16, lane, aq);

  v8f accO[8];
  const v8f vzero = {};
#pragma unroll
  for (int dt = 0; dt < 8; ++dt) accO[dt] = vzero;
  float mrow[8], lrow[8];
#pragma unroll
  for (int r = 0; r < 8; ++r) { mrow[r] = FNEG_MAX; lrow[r] = 0.f; }

  __bf16* Pw = Pl + w * 16 * PST;

  for (int kb = 0; kb < NKB; ++kb) {
    __syncthreads();
    stageK(Kl, kg + (size_t)kb * BC * DIM, tid);
    stageVt(Vt, vg + (size_t)kb * BC * DIM, tid);
    __syncthreads();
    __builtin_prefetch(kg + (size_t)((kb + 1) & (NKB - 1)) * BC * DIM + tid * 16, 0, 1);
    __builtin_prefetch(vg + (size_t)((kb + 1) & (NKB - 1)) * BC * DIM + tid * 16, 0, 1);

    v8f sc[4];
    qkTiles(Kl, aq, lane, sc);

    // scale + quantile mask; row = r + 8*half, col = t*16 + rl
    float pv[4][8], rmax[8];
#pragma unroll
    for (int r = 0; r < 8; ++r) rmax[r] = FNEG_MAX;
#pragma unroll
    for (int t = 0; t < 4; ++t) {
#pragma unroll
      for (int r = 0; r < 8; ++r) {
        float s = sc[t][r] * QK_SCALE;
        s = (s > thr) ? s : FNEG_MAX;
        pv[t][r] = s;
        rmax[r] = fmaxf(rmax[r], s);
      }
    }
    // row-max across the 16 lanes of each half-wave
#pragma unroll
    for (int r = 0; r < 8; ++r) {
#pragma unroll
      for (int mk = 1; mk < 16; mk <<= 1)
        rmax[r] = fmaxf(rmax[r], __shfl_xor(rmax[r], mk, 32));
    }
    float corr[8];
#pragma unroll
    for (int r = 0; r < 8; ++r) {
      const float mn = fmaxf(mrow[r], rmax[r]);
      corr[r] = __expf(mrow[r] - mn);
      mrow[r] = mn;
    }
    float rsum[8];
#pragma unroll
    for (int r = 0; r < 8; ++r) rsum[r] = 0.f;
#pragma unroll
    for (int t = 0; t < 4; ++t) {
#pragma unroll
      for (int r = 0; r < 8; ++r) {
        const float p = __expf(pv[t][r] - mrow[r]);
        rsum[r] += p;
        Pw[(r + 8 * half) * PST + t * 16 + rl] = bf16val(p);  // C-layout -> LDS
      }
    }
#pragma unroll
    for (int r = 0; r < 8; ++r) {
      float rs = rsum[r];
#pragma unroll
      for (int mk = 1; mk < 16; mk <<= 1) rs += __shfl_xor(rs, mk, 32);
      lrow[r] = lrow[r] * corr[r] + rs;
#pragma unroll
      for (int dt = 0; dt < 8; ++dt) accO[dt][r] *= corr[r];
    }

    // wave-private LDS transpose: writes above, reads below (in-order DS pipe)
    asm volatile("s_wait_dscnt 0" ::: "memory");

    // P as A-fragments (16 x 64 over two 32-wide K chunks)
    v16bf pa[2];
#pragma unroll
    for (int c2 = 0; c2 < 2; ++c2) {
      union { u32x4 qq[2]; v16bf vv; } u;
      const __bf16* pr = Pw + rl * PST + c2 * 32 + half * 16;
      u.qq[0] = *(const u32x4*)pr;          // n = c2*32 + half*16 + 0..7
      u.qq[1] = *(const u32x4*)(pr + 8);    // n = c2*32 + half*16 + 8..15
      pa[c2] = u.vv;
    }
    // O += P @ V : 8 d-tiles x 2 k-chunks of WMMA
#pragma unroll
    for (int dt = 0; dt < 8; ++dt) {
#pragma unroll
      for (int c2 = 0; c2 < 2; ++c2) {
        v16bf b = ldsB16x16(Vt + (dt * 16 + rl) * VST + c2 * 32 + half * 16);
        accO[dt] = __builtin_amdgcn_wmma_f32_16x16x32_bf16(false, pa[c2], false, b,
                                                           (short)0, accO[dt], false, false);
      }
    }
  }

  // epilogue: normalize and store fp32
  float* og = out + base + (size_t)(qb * BR + w * 16) * DIM;
#pragma unroll
  for (int dt = 0; dt < 8; ++dt) {
#pragma unroll
    for (int r = 0; r < 8; ++r) {
      og[(r + 8 * half) * DIM + dt * 16 + rl] = accO[dt][r] / lrow[r];
    }
  }
}

// ======================= host launcher =======================
extern "C" void kernel_launch(void* const* d_in, const int* in_sizes, int n_in,
                              void* d_out, int out_size, void* d_ws, size_t ws_size,
                              hipStream_t stream) {
  const float* q = (const float*)d_in[0];
  const float* k = (const float*)d_in[1];
  const float* v = (const float*)d_in[2];
  float* out = (float*)d_out;

  const size_t need = (size_t)NBINS * sizeof(unsigned int) + sizeof(float);
  const int nblk = NB_B * NB_H * NQB;  // 1024 workgroups

  if (d_ws && ws_size >= need) {
    unsigned int* hist = (unsigned int*)d_ws;
    float* thr = (float*)((char*)d_ws + (size_t)NBINS * sizeof(unsigned int));
    psi_zero_kernel<<<(NBINS + 255) / 256, 256, 0, stream>>>(hist, NBINS);
    psi_hist_kernel<<<nblk, 128, 0, stream>>>(q, k, hist);
    psi_select_kernel<<<1, 32, 0, stream>>>(hist, thr);
    psi_flash_kernel<<<nblk, 128, 0, stream>>>(q, k, v, thr, out);
  } else {
    // workspace too small: scores ~ N(0,1), 0.5-quantile ~= 0.0
    psi_flash_kernel<<<nblk, 128, 0, stream>>>(q, k, v, nullptr, out);
  }
}
